// ModeAttention_62088047231246
// MI455X (gfx1250) — compile-verified, matrix operand order checked by reading, not run on hardware
//
#include <hip/hip_runtime.h>
#include <hip/hip_bf16.h>

// ---------------- problem constants (QCNet mode attention) ----------------
#define A_DIM   48
#define S_DIM   30
#define M_DIM   6
#define HID     128
#define NH      8
#define HD      16
#define NFB     64
#define NNODES  (A_DIM * S_DIM * M_DIM)   // 8640
#define NEB     10000                      // base edges
#define NEXP    (NEB * M_DIM * M_DIM)      // 360000 expanded edges
#define NTILES  (NEXP / 16)                // 22500
#define K1PAD   160                        // 129 padded to multiple of 32

#define PI_F     3.14159265358979f
#define TWO_PI_F 6.28318530717959f

typedef __attribute__((ext_vector_type(16))) _Float16 v16h;
typedef __attribute__((ext_vector_type(8)))  float    v8f;

// ---------------- WMMA helpers ----------------
// 16x32 f16 fragment loader (row-major, stride in halves, 16B aligned rows).
// Per-lane layout for V_WMMA_F32_16X16X32_F16 operands:
//   lane<16 : row = lane,    K = {0..7, 16..23}
//   lane>=16: row = lane-16, K = {8..15, 24..31}
__device__ inline v16h load_frag(const _Float16* p0, int stride) {
  int lane = threadIdx.x & 31;
  const _Float16* p = p0 + (size_t)(lane & 15) * stride + ((lane >> 4) << 3);
  union { uint4 u[2]; v16h h; } t;
  t.u[0] = *(const uint4*)(p);
  t.u[1] = *(const uint4*)(p + 16);
  return t.h;
}

__device__ inline v8f wmma32(v16h a, v16h b, v8f c) {
  // D = A(16x32) x B(32x16) + C, f32 accumulate
  return __builtin_amdgcn_wmma_f32_16x16x32_f16(
      false, a, false, b, (short)0, c, false, false);
}

// ordered-int trick for float atomic max (works for mixed signs)
__device__ inline void atomicMaxF32(float* addr, float val) {
  int iv = __float_as_int(val);
  if (iv >= 0) atomicMax((int*)addr, iv);
  else         atomicMin((unsigned int*)addr, (unsigned int)iv);
}

// ---------------- small utility kernels ----------------
__global__ void cvt_f16_kernel(const float* __restrict__ in,
                               _Float16* __restrict__ out, int n) {
  int i = blockIdx.x * blockDim.x + threadIdx.x;
  if (i < n) out[i] = (_Float16)in[i];
}

// w1 [3][128][129] -> f16 padded [3][128][160] (zeros in tail)
__global__ void cvt_w1_pad_kernel(const float* __restrict__ w1,
                                  _Float16* __restrict__ w1p) {
  int i = blockIdx.x * blockDim.x + threadIdx.x;
  if (i >= 3 * HID * K1PAD) return;
  int j  = i % K1PAD;
  int co = i / K1PAD;
  float v = (j < 2 * NFB + 1) ? w1[(size_t)co * (2 * NFB + 1) + j] : 0.f;
  w1p[i] = (_Float16)v;
}

// xf[node][c] = x[a][m][s][c] + mode_emb[m][c] + time_emb[pred][c],
// node = (a*S+s)*M+m
__global__ void prep_nodes_kernel(const float* __restrict__ x,
                                  const float* __restrict__ mode_emb,
                                  const float* __restrict__ time_emb,
                                  const int* __restrict__ pred,
                                  float* __restrict__ xf) {
  int i = blockIdx.x * blockDim.x + threadIdx.x;
  if (i >= NNODES * HID) return;
  int c = i & 127;
  int n = i >> 7;
  int m  = n % M_DIM;
  int as = n / M_DIM;
  int s  = as % S_DIM;
  int a  = as / S_DIM;
  int ps = pred[0];
  xf[i] = x[(size_t)(((a * M_DIM + m) * S_DIM + s)) * HID + c]
        + mode_emb[(size_t)m * HID + c]
        + time_emb[(size_t)ps * HID + c];
}

__global__ void init_bufs_kernel(float* mmax, float* denom, float* agg) {
  int i = blockIdx.x * blockDim.x + threadIdx.x;
  if (i < NNODES * NH) { ((int*)mmax)[i] = 0xFF800000; denom[i] = 0.f; }
  if (i < NNODES * HID) agg[i] = 0.f;
}

// LayerNorm over 128 features, optional residual add: out = (res?) + LN(in)*g+b
__global__ __launch_bounds__(128) void ln128_kernel(
    const float* __restrict__ in, const float* __restrict__ res,
    const float* __restrict__ g, const float* __restrict__ b,
    float* __restrict__ out) {
  __shared__ float sbuf[128];
  int row = blockIdx.x;
  int t   = threadIdx.x;
  float x = in[(size_t)row * HID + t];
  sbuf[t] = x; __syncthreads();
  for (int s = 64; s > 0; s >>= 1) { if (t < s) sbuf[t] += sbuf[t + s]; __syncthreads(); }
  float mu = sbuf[0] * (1.f / 128.f);
  __syncthreads();
  float d = x - mu;
  sbuf[t] = d * d; __syncthreads();
  for (int s = 64; s > 0; s >>= 1) { if (t < s) sbuf[t] += sbuf[t + s]; __syncthreads(); }
  float inv = rsqrtf(sbuf[0] * (1.f / 128.f) + 1e-5f);
  float y = d * inv * g[t] + b[t];
  if (res) y += res[(size_t)row * HID + t];
  out[(size_t)row * HID + t] = y;
}

// final: d_out[a][m][s][c] = x1 + LN(ff2)*g+b   (fused transpose scatter)
__global__ __launch_bounds__(128) void final_ln_scatter_kernel(
    const float* __restrict__ ff2, const float* __restrict__ x1,
    const float* __restrict__ g, const float* __restrict__ b,
    float* __restrict__ out) {
  __shared__ float sbuf[128];
  int row = blockIdx.x;
  int t   = threadIdx.x;
  float x = ff2[(size_t)row * HID + t];
  sbuf[t] = x; __syncthreads();
  for (int s = 64; s > 0; s >>= 1) { if (t < s) sbuf[t] += sbuf[t + s]; __syncthreads(); }
  float mu = sbuf[0] * (1.f / 128.f);
  __syncthreads();
  float d = x - mu;
  sbuf[t] = d * d; __syncthreads();
  for (int s = 64; s > 0; s >>= 1) { if (t < s) sbuf[t] += sbuf[t + s]; __syncthreads(); }
  float inv = rsqrtf(sbuf[0] * (1.f / 128.f) + 1e-5f);
  float y = x1[(size_t)row * HID + t] + d * inv * g[t] + b[t];
  int m  = row % M_DIM;
  int as = row / M_DIM;
  int s_ = as % S_DIM;
  int a  = as / S_DIM;
  out[(size_t)(((a * M_DIM + m) * S_DIM + s_)) * HID + t] = y;
}

// cat[n][0..127]=agg, cat[n][128..255]=xn  (f16)
__global__ void build_cat_kernel(const float* __restrict__ agg,
                                 const float* __restrict__ xn,
                                 _Float16* __restrict__ cat) {
  int i = blockIdx.x * blockDim.x + threadIdx.x;
  if (i >= NNODES * 256) return;
  int j = i & 255;
  int n = i >> 8;
  float v = (j < 128) ? agg[(size_t)n * HID + j] : xn[(size_t)n * HID + (j - 128)];
  cat[i] = (_Float16)v;
}

// o = agg + g*(s - agg), stored f16 (only consumed by to_out GEMM)
__global__ void gate_combine_kernel(const float* __restrict__ agg,
                                    const float* __restrict__ gbuf,
                                    const float* __restrict__ sv,
                                    _Float16* __restrict__ out) {
  int i = blockIdx.x * blockDim.x + threadIdx.x;
  if (i >= NNODES * HID) return;
  float a = agg[i];
  out[i] = (_Float16)(a + gbuf[i] * (sv[i] - a));
}

// ---------------- generic wave-per-tile WMMA GEMM ----------------
// C[M x N] = A[M x K](f16) * W[N x K]^T(f16) + bias ; act: 0 none, 1 relu, 2 sigmoid
__global__ __launch_bounds__(32) void gemm16_kernel(
    const _Float16* __restrict__ Am, const _Float16* __restrict__ Wm,
    const float* __restrict__ bias, float* __restrict__ C,
    int K, int N, int act) {
  int mt = blockIdx.x, nt = blockIdx.y;
  int lane = threadIdx.x & 31;
  v8f acc = {};
  const _Float16* ap = Am + (size_t)mt * 16 * K;
  const _Float16* wp = Wm + (size_t)nt * 16 * K;
  for (int k0 = 0; k0 < K; k0 += 32)
    acc = wmma32(load_frag(ap + k0, K), load_frag(wp + k0, K), acc);
  int n  = nt * 16 + (lane & 15);
  float bv = bias ? bias[n] : 0.f;
  int m0 = mt * 16 + ((lane >> 4) << 3);
#pragma unroll
  for (int i = 0; i < 8; ++i) {
    float v = acc[i] + bv;
    if (act == 1)      v = fmaxf(v, 0.f);
    else if (act == 2) v = 1.f / (1.f + __expf(-v));
    C[(size_t)(m0 + i) * N + n] = v;
  }
}

// ---------------- fused edge kernel: fourier emb + pre_r LN + k_r + sim ----
struct EdgeArgs {
  const int* edges;
  const float* pos;
  const float* head;
  const float* freqs;
  const _Float16* w1p;      // [3][128][160]
  const float* b1;          // [3][128]
  const float* ln1_g; const float* ln1_b;
  const _Float16* w2h;      // [3][128][128]
  const float* b2;
  const float* out_ln_g; const float* out_ln_b;
  const _Float16* out_wh;   // [128][128]
  const float* out_b;
  const float* pre_r_g; const float* pre_r_b;
  const _Float16* to_kr_wh; // [128][128]
  const float* q; const float* k;   // [NNODES][128]
  _Float16* rn_out;         // [NEXP][128]
  float* sim;               // [NEXP][8]
  float* mmax;              // [NNODES][8]
};

__global__ __launch_bounds__(32) void edge_fourier_sim_kernel(EdgeArgs P) {
  __shared__ alignas(16) _Float16 feat[16][K1PAD];
  __shared__ alignas(16) float    h1[16][HID];
  __shared__ alignas(16) _Float16 h1h[16][HID];
  __shared__ alignas(16) float    hsum[16][HID];
  __shared__ int   s_src[16];
  __shared__ int   s_dst[16];
  __shared__ float s_rel[16][3];

  const int lane = threadIdx.x & 31;
  const int g0   = blockIdx.x * 16;

  if (lane < 16) {
    int g  = g0 + lane;
    int be = g / 36, p = g - be * 36;
    int mi = p % M_DIM, mj = p / M_DIM;
    int bs = P.edges[be];
    int bd = P.edges[NEB + be];
    s_src[lane] = bs * M_DIM + mi;
    s_dst[lane] = bd * M_DIM + mj;
    int as_ = bs / S_DIM, ss = bs - as_ * S_DIM;
    int ad_ = bd / S_DIM, sd = bd - ad_ * S_DIM;
    int is = (as_ * M_DIM + mi) * S_DIM + ss;
    int id = (ad_ * M_DIM + mj) * S_DIM + sd;
    float px = P.pos[(size_t)is * 2]     - P.pos[(size_t)id * 2];
    float py = P.pos[(size_t)is * 2 + 1] - P.pos[(size_t)id * 2 + 1];
    float hs = P.head[is], hd = P.head[id];
    float dist = sqrtf(px * px + py * py);
    float cx = __cosf(hd), cy = __sinf(hd);
    float dir = atan2f(cx * py - cy * px, cx * px + cy * py);
    float t = fmodf(hs - hd + PI_F, TWO_PI_F);
    if (t < 0.f) t += TWO_PI_F;
    s_rel[lane][0] = dist;
    s_rel[lane][1] = dir;
    s_rel[lane][2] = t - PI_F;
  }
  __syncthreads();

  for (int c = 0; c < 3; ++c) {
    // build fourier feature tile (cos | sin | raw | zero pad)
    for (int idx = lane; idx < 16 * K1PAD; idx += 32) {
      int e = idx / K1PAD, j = idx - e * K1PAD;
      float r = s_rel[e][c];
      float v;
      if (j < NFB)          v = __cosf(r * P.freqs[c * NFB + j] * TWO_PI_F);
      else if (j < 2 * NFB) v = __sinf(r * P.freqs[c * NFB + j - NFB] * TWO_PI_F);
      else                  v = (j == 2 * NFB) ? r : 0.f;
      feat[e][j] = (_Float16)v;
    }
    __syncthreads();
    // h1 = feat @ w1[c]^T + b1[c]   (K=160, 5 WMMA k-steps x 8 n-tiles)
    for (int nt = 0; nt < 8; ++nt) {
      v8f acc = {};
#pragma unroll
      for (int k0 = 0; k0 < K1PAD; k0 += 32)
        acc = wmma32(load_frag(&feat[0][0] + k0, K1PAD),
                     load_frag(P.w1p + (size_t)(c * HID + nt * 16) * K1PAD + k0, K1PAD),
                     acc);
      int n = nt * 16 + (lane & 15);
      float bv = P.b1[c * HID + n];
      int mb = (lane >> 4) << 3;
#pragma unroll
      for (int i = 0; i < 8; ++i) h1[mb + i][n] = acc[i] + bv;
    }
    __syncthreads();
    // LN + ReLU -> f16
    if (lane < 16) {
      int r = lane;
      float mu = 0.f;
      for (int j = 0; j < HID; ++j) mu += h1[r][j];
      mu *= (1.f / 128.f);
      float var = 0.f;
      for (int j = 0; j < HID; ++j) { float d = h1[r][j] - mu; var += d * d; }
      float inv = rsqrtf(var * (1.f / 128.f) + 1e-5f);
      for (int j = 0; j < HID; ++j) {
        float y = (h1[r][j] - mu) * inv * P.ln1_g[c * HID + j] + P.ln1_b[c * HID + j];
        h1h[r][j] = (_Float16)fmaxf(y, 0.f);
      }
    }
    __syncthreads();
    // hsum += h1h @ w2[c]^T + b2[c]
    for (int nt = 0; nt < 8; ++nt) {
      v8f acc = {};
#pragma unroll
      for (int k0 = 0; k0 < HID; k0 += 32)
        acc = wmma32(load_frag(&h1h[0][0] + k0, HID),
                     load_frag(P.w2h + (size_t)(c * HID + nt * 16) * HID + k0, HID),
                     acc);
      int n = nt * 16 + (lane & 15);
      float bv = P.b2[c * HID + n];
      int mb = (lane >> 4) << 3;
#pragma unroll
      for (int i = 0; i < 8; ++i) {
        float prev = (c == 0) ? 0.f : hsum[mb + i][n];
        hsum[mb + i][n] = prev + acc[i] + bv;
      }
    }
    __syncthreads();
  }

  // out LN + ReLU
  if (lane < 16) {
    int r = lane;
    float mu = 0.f;
    for (int j = 0; j < HID; ++j) mu += hsum[r][j];
    mu *= (1.f / 128.f);
    float var = 0.f;
    for (int j = 0; j < HID; ++j) { float d = hsum[r][j] - mu; var += d * d; }
    float inv = rsqrtf(var * (1.f / 128.f) + 1e-5f);
    for (int j = 0; j < HID; ++j) {
      float y = (hsum[r][j] - mu) * inv * P.out_ln_g[j] + P.out_ln_b[j];
      h1h[r][j] = (_Float16)fmaxf(y, 0.f);
    }
  }
  __syncthreads();
  // r = h1h @ out_w^T + out_b -> h1 (f32)
  for (int nt = 0; nt < 8; ++nt) {
    v8f acc = {};
#pragma unroll
    for (int k0 = 0; k0 < HID; k0 += 32)
      acc = wmma32(load_frag(&h1h[0][0] + k0, HID),
                   load_frag(P.out_wh + (size_t)(nt * 16) * HID + k0, HID), acc);
    int n = nt * 16 + (lane & 15);
    float bv = P.out_b[n];
    int mb = (lane >> 4) << 3;
#pragma unroll
    for (int i = 0; i < 8; ++i) h1[mb + i][n] = acc[i] + bv;
  }
  __syncthreads();
  // pre_r LN -> rn (f16 in LDS + spill to global for the aggregation pass)
  if (lane < 16) {
    int r = lane;
    float mu = 0.f;
    for (int j = 0; j < HID; ++j) mu += h1[r][j];
    mu *= (1.f / 128.f);
    float var = 0.f;
    for (int j = 0; j < HID; ++j) { float d = h1[r][j] - mu; var += d * d; }
    float inv = rsqrtf(var * (1.f / 128.f) + 1e-5f);
    for (int j = 0; j < HID; ++j) {
      float y = (h1[r][j] - mu) * inv * P.pre_r_g[j] + P.pre_r_b[j];
      _Float16 hv = (_Float16)y;
      h1h[r][j] = hv;
      P.rn_out[(size_t)(g0 + r) * HID + j] = hv;
    }
  }
  __syncthreads();
  // kr = rn @ to_k_r^T (no bias) -> hsum
  for (int nt = 0; nt < 8; ++nt) {
    v8f acc = {};
#pragma unroll
    for (int k0 = 0; k0 < HID; k0 += 32)
      acc = wmma32(load_frag(&h1h[0][0] + k0, HID),
                   load_frag(P.to_kr_wh + (size_t)(nt * 16) * HID + k0, HID), acc);
    int n = nt * 16 + (lane & 15);
    int mb = (lane >> 4) << 3;
#pragma unroll
    for (int i = 0; i < 8; ++i) hsum[mb + i][n] = acc[i];
  }
  __syncthreads();
  // sim[e][h] = (q[dst] . (k[src]+kr)) * HD^-0.5 ; atomic segment-max
#pragma unroll
  for (int t = 0; t < 4; ++t) {
    int idx = t * 32 + lane;
    int e = idx >> 3, h = idx & 7;
    int src = s_src[e], dst = s_dst[e];
    float sum = 0.f;
#pragma unroll
    for (int d = 0; d < HD; ++d) {
      int col = h * HD + d;
      float kj = P.k[(size_t)src * HID + col] + hsum[e][col];
      sum += P.q[(size_t)dst * HID + col] * kj;
    }
    sum *= 0.25f;   // HD^-0.5
    P.sim[(size_t)(g0 + e) * NH + h] = sum;
    atomicMaxF32(&P.mmax[dst * NH + h], sum);
  }
}

// ---------------- softmax normalize (pass 2) ----------------
__global__ void softmax_norm_kernel(const int* __restrict__ edges,
                                    float* __restrict__ sim,
                                    const float* __restrict__ mmax,
                                    float* __restrict__ denom) {
  int i = blockIdx.x * blockDim.x + threadIdx.x;
  if (i >= NEXP * NH) return;
  int h = i & 7;
  int g = i >> 3;
  int be = g / 36, p = g - be * 36;
  int dst = edges[NEB + be] * M_DIM + p / M_DIM;
  float ex = __expf(sim[i] - mmax[dst * NH + h]);
  sim[i] = ex;
  atomicAdd(&denom[dst * NH + h], ex);
}

// ---------------- aggregation (pass 3): v_r WMMA + weighted scatter-add ----
struct AggArgs {
  const int* edges;
  const _Float16* rn;       // [NEXP][128]
  const _Float16* to_vr_wh; // [128][128]
  const float* to_vr_b;
  const float* v;           // [NNODES][128]
  const float* sim;         // holds exp() after pass 2
  const float* denom;
  float* agg;               // [NNODES][128]
};

__global__ __launch_bounds__(32) void edge_aggregate_kernel(AggArgs P) {
  __shared__ alignas(16) float vr[16][HID];
  __shared__ int s_src[16];
  __shared__ int s_dst[16];
  int lane = threadIdx.x & 31;
  int g0   = blockIdx.x * 16;
  if (lane < 16) {
    int g = g0 + lane;
    int be = g / 36, p = g - be * 36;
    s_src[lane] = P.edges[be] * M_DIM + p % M_DIM;
    s_dst[lane] = P.edges[NEB + be] * M_DIM + p / M_DIM;
  }
  __syncthreads();
  const _Float16* rp = P.rn + (size_t)g0 * HID;
  for (int nt = 0; nt < 8; ++nt) {
    v8f acc = {};
#pragma unroll
    for (int k0 = 0; k0 < HID; k0 += 32)
      acc = wmma32(load_frag(rp + k0, HID),
                   load_frag(P.to_vr_wh + (size_t)(nt * 16) * HID + k0, HID), acc);
    int n = nt * 16 + (lane & 15);
    float bv = P.to_vr_b[n];
    int mb = (lane >> 4) << 3;
#pragma unroll
    for (int i = 0; i < 8; ++i) vr[mb + i][n] = acc[i] + bv;
  }
  __syncthreads();
  for (int t = 0; t < 64; ++t) {
    int idx = t * 32 + lane;
    int e = idx >> 7, col = idx & 127, h = col >> 4;
    int src = s_src[e], dst = s_dst[e];
    float w  = P.sim[(size_t)(g0 + e) * NH + h] / P.denom[dst * NH + h];
    float vj = P.v[(size_t)src * HID + col] + vr[e][col];
    atomicAdd(&P.agg[(size_t)dst * HID + col], w * vj);
  }
}

// ---------------- host-side launch ----------------
extern "C" void kernel_launch(void* const* d_in, const int* in_sizes, int n_in,
                              void* d_out, int out_size, void* d_ws, size_t ws_size,
                              hipStream_t stream) {
  (void)in_sizes; (void)n_in; (void)out_size; (void)ws_size;
  const float* x        = (const float*)d_in[0];
  const float* pos      = (const float*)d_in[1];
  const float* head     = (const float*)d_in[2];
  const int*   edges    = (const int*)  d_in[3];
  const float* mode_emb = (const float*)d_in[4];
  const float* time_emb = (const float*)d_in[5];
  const float* freqs    = (const float*)d_in[6];
  const float* w1       = (const float*)d_in[7];
  const float* b1       = (const float*)d_in[8];
  const float* ln1_g    = (const float*)d_in[9];
  const float* ln1_b    = (const float*)d_in[10];
  const float* w2       = (const float*)d_in[11];
  const float* b2       = (const float*)d_in[12];
  const float* out_ln_g = (const float*)d_in[13];
  const float* out_ln_b = (const float*)d_in[14];
  const float* out_w    = (const float*)d_in[15];
  const float* out_b    = (const float*)d_in[16];
  const float* pre_x_g  = (const float*)d_in[17];
  const float* pre_x_b  = (const float*)d_in[18];
  const float* pre_r_g  = (const float*)d_in[19];
  const float* pre_r_b  = (const float*)d_in[20];
  const float* to_q_w   = (const float*)d_in[21];
  const float* to_q_b   = (const float*)d_in[22];
  const float* to_k_w   = (const float*)d_in[23];
  const float* to_v_w   = (const float*)d_in[24];
  const float* to_v_b   = (const float*)d_in[25];
  const float* to_k_r_w = (const float*)d_in[26];
  const float* to_v_r_w = (const float*)d_in[27];
  const float* to_v_r_b = (const float*)d_in[28];
  const float* to_s_w   = (const float*)d_in[29];
  const float* to_s_b   = (const float*)d_in[30];
  const float* to_g_w   = (const float*)d_in[31];
  const float* to_g_b   = (const float*)d_in[32];
  const float* to_out_w = (const float*)d_in[33];
  const float* to_out_b = (const float*)d_in[34];
  const float* post_g   = (const float*)d_in[35];
  const float* post_b   = (const float*)d_in[36];
  const float* ff_pre_g = (const float*)d_in[37];
  const float* ff_pre_b = (const float*)d_in[38];
  const float* ff_w1    = (const float*)d_in[39];
  const float* ff_b1    = (const float*)d_in[40];
  const float* ff_w2    = (const float*)d_in[41];
  const float* ff_b2    = (const float*)d_in[42];
  const float* ff_post_g= (const float*)d_in[43];
  const float* ff_post_b= (const float*)d_in[44];
  const int*   pred     = (const int*)  d_in[45];

  char* wp = (char*)d_ws;
  auto alloc = [&](size_t bytes) -> void* {
    void* r = (void*)wp;
    wp += (bytes + 255) & ~(size_t)255;
    return r;
  };
  float*    xf     = (float*)   alloc((size_t)NNODES * HID * 4);
  float*    xn     = (float*)   alloc((size_t)NNODES * HID * 4);
  _Float16* xn_h   = (_Float16*)alloc((size_t)NNODES * HID * 2);
  float*    qb     = (float*)   alloc((size_t)NNODES * HID * 4);
  float*    kb     = (float*)   alloc((size_t)NNODES * HID * 4);
  float*    vb     = (float*)   alloc((size_t)NNODES * HID * 4);
  float*    sb     = (float*)   alloc((size_t)NNODES * HID * 4);
  float*    mmax   = (float*)   alloc((size_t)NNODES * NH * 4);
  float*    denom  = (float*)   alloc((size_t)NNODES * NH * 4);
  float*    agg    = (float*)   alloc((size_t)NNODES * HID * 4);
  float*    sim    = (float*)   alloc((size_t)NEXP * NH * 4);
  _Float16* rn_h   = (_Float16*)alloc((size_t)NEXP * HID * 2);
  _Float16* cat_h  = (_Float16*)alloc((size_t)NNODES * 256 * 2);
  float*    gbuf   = (float*)   alloc((size_t)NNODES * HID * 4);
  _Float16* obuf_h = (_Float16*)alloc((size_t)NNODES * HID * 2);
  float*    o2     = (float*)   alloc((size_t)NNODES * HID * 4);
  float*    x1     = (float*)   alloc((size_t)NNODES * HID * 4);
  float*    ffin   = (float*)   alloc((size_t)NNODES * HID * 4);
  _Float16* ffin_h = (_Float16*)alloc((size_t)NNODES * HID * 2);
  float*    ff1    = (float*)   alloc((size_t)NNODES * 512 * 4);
  _Float16* ff1_h  = (_Float16*)alloc((size_t)NNODES * 512 * 2);
  float*    ff2    = (float*)   alloc((size_t)NNODES * HID * 4);
  // f16 weight copies
  _Float16* w1p      = (_Float16*)alloc((size_t)3 * HID * K1PAD * 2);
  _Float16* w2h      = (_Float16*)alloc((size_t)3 * HID * HID * 2);
  _Float16* out_wh   = (_Float16*)alloc((size_t)HID * HID * 2);
  _Float16* to_q_wh  = (_Float16*)alloc((size_t)HID * HID * 2);
  _Float16* to_k_wh  = (_Float16*)alloc((size_t)HID * HID * 2);
  _Float16* to_v_wh  = (_Float16*)alloc((size_t)HID * HID * 2);
  _Float16* to_s_wh  = (_Float16*)alloc((size_t)HID * HID * 2);
  _Float16* to_kr_wh = (_Float16*)alloc((size_t)HID * HID * 2);
  _Float16* to_vr_wh = (_Float16*)alloc((size_t)HID * HID * 2);
  _Float16* to_out_wh= (_Float16*)alloc((size_t)HID * HID * 2);
  _Float16* to_g_wh  = (_Float16*)alloc((size_t)HID * 256 * 2);
  _Float16* ff_w1h   = (_Float16*)alloc((size_t)512 * HID * 2);
  _Float16* ff_w2h   = (_Float16*)alloc((size_t)HID * 512 * 2);

  auto cvt = [&](const float* src, _Float16* dst, int n) {
    cvt_f16_kernel<<<(n + 255) / 256, 256, 0, stream>>>(src, dst, n);
  };

  // weight conversions (deterministic, redone every launch)
  cvt_w1_pad_kernel<<<(3 * HID * K1PAD + 255) / 256, 256, 0, stream>>>(w1, w1p);
  cvt(w2, w2h, 3 * HID * HID);
  cvt(out_w, out_wh, HID * HID);
  cvt(to_q_w, to_q_wh, HID * HID);
  cvt(to_k_w, to_k_wh, HID * HID);
  cvt(to_v_w, to_v_wh, HID * HID);
  cvt(to_s_w, to_s_wh, HID * HID);
  cvt(to_k_r_w, to_kr_wh, HID * HID);
  cvt(to_v_r_w, to_vr_wh, HID * HID);
  cvt(to_out_w, to_out_wh, HID * HID);
  cvt(to_g_w, to_g_wh, HID * 256);
  cvt(ff_w1, ff_w1h, 512 * HID);
  cvt(ff_w2, ff_w2h, HID * 512);

  // node features + prenorm + q/k/v/s projections
  prep_nodes_kernel<<<(NNODES * HID + 255) / 256, 256, 0, stream>>>(
      x, mode_emb, time_emb, pred, xf);
  ln128_kernel<<<NNODES, 128, 0, stream>>>(xf, nullptr, pre_x_g, pre_x_b, xn);
  cvt(xn, xn_h, NNODES * HID);
  dim3 g128(NNODES / 16, HID / 16);
  gemm16_kernel<<<g128, 32, 0, stream>>>(xn_h, to_q_wh, to_q_b, qb, HID, HID, 0);
  gemm16_kernel<<<g128, 32, 0, stream>>>(xn_h, to_k_wh, nullptr, kb, HID, HID, 0);
  gemm16_kernel<<<g128, 32, 0, stream>>>(xn_h, to_v_wh, to_v_b, vb, HID, HID, 0);
  gemm16_kernel<<<g128, 32, 0, stream>>>(xn_h, to_s_wh, to_s_b, sb, HID, HID, 0);

  init_bufs_kernel<<<(NNODES * HID + 255) / 256, 256, 0, stream>>>(mmax, denom, agg);

  // pass 1: fused fourier MLP + k_r + scores + segment max
  EdgeArgs ea;
  ea.edges = edges; ea.pos = pos; ea.head = head; ea.freqs = freqs;
  ea.w1p = w1p; ea.b1 = b1; ea.ln1_g = ln1_g; ea.ln1_b = ln1_b;
  ea.w2h = w2h; ea.b2 = b2; ea.out_ln_g = out_ln_g; ea.out_ln_b = out_ln_b;
  ea.out_wh = out_wh; ea.out_b = out_b; ea.pre_r_g = pre_r_g; ea.pre_r_b = pre_r_b;
  ea.to_kr_wh = to_kr_wh; ea.q = qb; ea.k = kb;
  ea.rn_out = rn_h; ea.sim = sim; ea.mmax = mmax;
  edge_fourier_sim_kernel<<<NTILES, 32, 0, stream>>>(ea);

  // pass 2: softmax normalize
  softmax_norm_kernel<<<(NEXP * NH + 255) / 256, 256, 0, stream>>>(
      edges, sim, mmax, denom);

  // pass 3: v_r + weighted aggregation
  AggArgs aa;
  aa.edges = edges; aa.rn = rn_h; aa.to_vr_wh = to_vr_wh; aa.to_vr_b = to_v_r_b;
  aa.v = vb; aa.sim = sim; aa.denom = denom; aa.agg = agg;
  edge_aggregate_kernel<<<NTILES, 32, 0, stream>>>(aa);

  // node epilogue: gating, out proj, residual LNs, FFN
  build_cat_kernel<<<(NNODES * 256 + 255) / 256, 256, 0, stream>>>(agg, xn, cat_h);
  gemm16_kernel<<<g128, 32, 0, stream>>>(cat_h, to_g_wh, to_g_b, gbuf, 256, HID, 2);
  gate_combine_kernel<<<(NNODES * HID + 255) / 256, 256, 0, stream>>>(
      agg, gbuf, sb, obuf_h);
  gemm16_kernel<<<g128, 32, 0, stream>>>(obuf_h, to_out_wh, to_out_b, o2, HID, HID, 0);
  ln128_kernel<<<NNODES, 128, 0, stream>>>(o2, xf, post_g, post_b, x1);
  ln128_kernel<<<NNODES, 128, 0, stream>>>(x1, nullptr, ff_pre_g, ff_pre_b, ffin);
  cvt(ffin, ffin_h, NNODES * HID);
  dim3 g512(NNODES / 16, 512 / 16);
  gemm16_kernel<<<g512, 32, 0, stream>>>(ffin_h, ff_w1h, ff_b1, ff1, HID, 512, 1);
  cvt(ff1, ff1_h, NNODES * 512);
  gemm16_kernel<<<g128, 32, 0, stream>>>(ff1_h, ff_w2h, ff_b2, ff2, 512, HID, 0);
  final_ln_scatter_kernel<<<NNODES, 128, 0, stream>>>(
      ff2, x1, ff_post_g, ff_post_b, (float*)d_out);
}
